// LinearAttention_352187319130
// MI455X (gfx1250) — matrix operand order repaired and checked by reading, not compiled, not run on hardware
//
#include <hip/hip_runtime.h>
#include <hip/hip_bf16.h>

// ---------------------------------------------------------------------------
// Linear attention, MI455X (gfx1250, wave32, WMMA + async-to-LDS).
// Per batch: qkv GEMM (WMMA f16) -> k-row softmax -> context partials (WMMA)
//            -> fold into G (incl. W_out + 1/rowsum) -> out GEMM (WMMA) + bias
// ---------------------------------------------------------------------------

typedef _Float16 half_t;
typedef __attribute__((ext_vector_type(16))) _Float16 v16h;
typedef __attribute__((ext_vector_type(8)))  _Float16 v8h;
typedef __attribute__((ext_vector_type(2)))  _Float16 v2h;
typedef __attribute__((ext_vector_type(8)))  float    v8f;

#define NTOK    16384      // 128*128 spatial positions
#define CH      128        // DIM (= K of both big GEMMs)
#define NHEAD   4
#define QK_M    384        // 3 * HIDDEN
#define NBATCH  16
#define SCALE_Q 0.17677669529663687f   // 32^-0.5
#define LDSB_PITCH 136     // halves per LDS row: 128 K + 8 pad (272B = 17*16B)

// Stage B tile global->LDS with the CDNA5 async copy engine (ASYNCcnt path).
#define USE_ASYNC_STAGE 1

// ---------------- small f32 -> f16 convert ----------------
__global__ void convert_f32_to_f16_kernel(const float* __restrict__ src,
                                          half_t* __restrict__ dst, int n) {
    int i = blockIdx.x * 256 + threadIdx.x;
    if (i < n) dst[i] = (half_t)src[i];
}

// ---------------- WMMA GEMM: C[MxN] = A16[MxK=128] @ B[128xN] ----------------
// mode 0: qkv  (rows <128 scaled by qscale; rows >=256 mirrored to v16out f16)
// mode 1: out  (adds bias[o])
// Block = 256 thr = 8 waves; wg tile 64(M) x 128(N); wave tile 16 x 64.
// Entire K=128 B-panel staged once into transposed f16 LDS; K-loop barrier-free.
__global__ __launch_bounds__(256) void gemm_wmma_kernel(
    const half_t* __restrict__ A16, const float* __restrict__ B,
    float* __restrict__ Cout, half_t* __restrict__ v16out,
    const float* __restrict__ bias,
    int M, int N, int mode, float qscale)
{
    __shared__ __align__(16) half_t ldsB[128 * LDSB_PITCH];   // [n][k] f16, 34.8KB
#if USE_ASYNC_STAGE
    __shared__ __align__(16) float rawB[32 * 128];            // raw f32 chunk, 16KB
#endif

    const int tid   = threadIdx.x;
    const int wave  = tid >> 5;
    const int lane  = tid & 31;
    const int waveM = wave >> 1;       // 0..3
    const int waveN = wave & 1;        // 0..1
    const int mBase = blockIdx.y * 64;
    const int nBase = blockIdx.x * 128;
    const int l15   = lane & 15;
    const int hi    = lane >> 4;       // half-wave select

    // ---------------- stage B[0:128][nBase:nBase+128] -> ldsB[n][k] f16 -------
#if USE_ASYNC_STAGE
    const unsigned rawOff = (unsigned)(size_t)(void*)rawB;  // LDS byte offset
    for (int c = 0; c < 4; ++c) {                 // 4 chunks of 32 k-rows
#pragma unroll
        for (int i = 0; i < 4; ++i) {             // 4 async b128 per lane
            int fid = tid + i * 256;              // 0..1023 float4 slots
            unsigned kkl = (unsigned)(fid >> 5);  // 0..31
            unsigned nn  = (unsigned)(fid & 31) * 4u;
            const float* gp = B + (unsigned)(c * 32 + kkl) * (unsigned)N
                                + (unsigned)nBase + nn;
            unsigned lo = rawOff + (unsigned)fid * 16u;
            asm volatile("global_load_async_to_lds_b128 %0, %1, off"
                         :: "v"(lo), "v"((unsigned long long)(size_t)gp)
                         : "memory");
        }
        asm volatile("s_wait_asynccnt 0x0" ::: "memory");
        __syncthreads();                          // chunk visible to all waves
        // transpose + cvt chunk (32 k) into ldsB
#pragma unroll
        for (int it = 0; it < 2; ++it) {
            int id  = tid + it * 256;             // 0..511
            int kkl = (id >> 5) * 2;              // even local k
            int nn  = (id & 31) * 4;
            float4 r0 = *(const float4*)&rawB[kkl * 128 + nn];
            float4 r1 = *(const float4*)&rawB[(kkl + 1) * 128 + nn];
            int kg = c * 32 + kkl;
            *(v2h*)&ldsB[(nn + 0) * LDSB_PITCH + kg] = (v2h){(half_t)r0.x, (half_t)r1.x};
            *(v2h*)&ldsB[(nn + 1) * LDSB_PITCH + kg] = (v2h){(half_t)r0.y, (half_t)r1.y};
            *(v2h*)&ldsB[(nn + 2) * LDSB_PITCH + kg] = (v2h){(half_t)r0.z, (half_t)r1.z};
            *(v2h*)&ldsB[(nn + 3) * LDSB_PITCH + kg] = (v2h){(half_t)r0.w, (half_t)r1.w};
        }
        __syncthreads();                          // rawB reusable next chunk
    }
#else
#pragma unroll
    for (int it = 0; it < 8; ++it) {
        int id  = tid + it * 256;                 // 0..2047
        int kk  = (id >> 5) * 2;                  // even k, 0..126
        int nn  = (id & 31) * 4;
        const float4 r0 = *(const float4*)&B[(unsigned)kk * (unsigned)N + (unsigned)(nBase + nn)];
        const float4 r1 = *(const float4*)&B[(unsigned)(kk + 1) * (unsigned)N + (unsigned)(nBase + nn)];
        *(v2h*)&ldsB[(nn + 0) * LDSB_PITCH + kk] = (v2h){(half_t)r0.x, (half_t)r1.x};
        *(v2h*)&ldsB[(nn + 1) * LDSB_PITCH + kk] = (v2h){(half_t)r0.y, (half_t)r1.y};
        *(v2h*)&ldsB[(nn + 2) * LDSB_PITCH + kk] = (v2h){(half_t)r0.z, (half_t)r1.z};
        *(v2h*)&ldsB[(nn + 3) * LDSB_PITCH + kk] = (v2h){(half_t)r0.w, (half_t)r1.w};
    }
    __syncthreads();
#endif

    // ---------------- barrier-free K loop: 4 x (2 A-loads, 8 ds, 4 WMMA) ------
    v8f acc[4];
#pragma unroll
    for (int t = 0; t < 4; ++t)
#pragma unroll
        for (int e = 0; e < 8; ++e) acc[t][e] = 0.0f;

    const int rowA = mBase + waveM * 16 + l15;    // A lane = M row (ISA layout)
    const half_t* aBase = A16 + (size_t)rowA * CH + hi * 8;

#pragma unroll
    for (int kc = 0; kc < 4; ++kc) {
        const int k0 = kc * 32;
        // A frag: i<8 -> K=k0+hi*8+i ; i>=8 -> K=k0+16+hi*8+(i-8)
        v8h a_lo = *(const v8h*)(aBase + k0);
        v8h a_hi = *(const v8h*)(aBase + k0 + 16);
        v16h afrag;
#pragma unroll
        for (int i = 0; i < 8; ++i) { afrag[i] = a_lo[i]; afrag[i + 8] = a_hi[i]; }

#pragma unroll
        for (int t = 0; t < 4; ++t) {             // B frag: lane col, K split 8/8 like A
            int nloc = waveN * 64 + t * 16 + l15;
            const half_t* bp = &ldsB[nloc * LDSB_PITCH + k0 + hi * 8];
            v8h b_lo = *(const v8h*)(bp);
            v8h b_hi = *(const v8h*)(bp + 16);
            v16h bfrag;
#pragma unroll
            for (int i = 0; i < 8; ++i) { bfrag[i] = b_lo[i]; bfrag[i + 8] = b_hi[i]; }
            acc[t] = __builtin_amdgcn_wmma_f32_16x16x32_f16(
                false, afrag, false, bfrag, (short)0, acc[t], false, false);
        }
    }

    // epilogue: D layout VGPR r -> M = r + (lane>=16 ? 8 : 0), N = lane&15
    const int mrow = mBase + waveM * 16 + (hi ? 8 : 0);
#pragma unroll
    for (int t = 0; t < 4; ++t) {
        int n = nBase + waveN * 64 + t * 16 + l15;
#pragma unroll
        for (int r = 0; r < 8; ++r) {
            int o = mrow + r;
            float val = acc[t][r];
            if (mode == 0) {
                if (o < 128) val *= qscale;                   // q rows
                Cout[(size_t)o * N + n] = val;
                if (o >= 256)                                 // v rows -> f16 mirror
                    v16out[(size_t)(o - 256) * N + n] = (half_t)val;
            } else {
                Cout[(size_t)o * N + n] = val + bias[o];
            }
        }
    }
}

// ---------------- row softmax stats over k: kexp f16, rowsum ----------------
__global__ __launch_bounds__(256) void softmax_rows_kernel(
    const float* __restrict__ kmat, half_t* __restrict__ kexp,
    float* __restrict__ rowsum)
{
    const int r   = blockIdx.x;          // 0..127
    const int tid = threadIdx.x;
    const float* row = kmat + (size_t)r * NTOK;
    __shared__ float red[256];

    float m = -1e30f;
    for (int j = tid; j < NTOK; j += 256) m = fmaxf(m, row[j]);
    red[tid] = m; __syncthreads();
    for (int s = 128; s > 0; s >>= 1) {
        if (tid < s) red[tid] = fmaxf(red[tid], red[tid + s]);
        __syncthreads();
    }
    const float rmax = red[0];
    __syncthreads();

    float sum = 0.0f;
    for (int j = tid; j < NTOK; j += 256) {
        float e = __expf(row[j] - rmax);
        half_t eh = (half_t)e;
        kexp[(size_t)r * NTOK + j] = eh;
        sum += (float)eh;                 // sum what the WMMA will actually see
    }
    red[tid] = sum; __syncthreads();
    for (int s = 128; s > 0; s >>= 1) {
        if (tid < s) red[tid] += red[tid + s];
        __syncthreads();
    }
    if (tid == 0) rowsum[r] = red[0];
}

// ---------------- context partials: ctxp[h*64+slot][d*32+e] ----------------
// grid = 32 (4 heads x 8 splits), 8 waves/wg; wave covers 256 n (K dim).
// Deterministic: every wave fully writes its own 1024-entry slot (no atomics).
__global__ __launch_bounds__(256) void context_kernel(
    const half_t* __restrict__ kexp, const half_t* __restrict__ v16,
    float* __restrict__ ctxp)
{
    const int head  = blockIdx.x >> 3;
    const int split = blockIdx.x & 7;
    const int wave  = threadIdx.x >> 5;
    const int lane  = threadIdx.x & 31;
    const int l15   = lane & 15;
    const int hi    = lane >> 4;

    v8f acc[2][2];
#pragma unroll
    for (int mt = 0; mt < 2; ++mt)
#pragma unroll
        for (int nt = 0; nt < 2; ++nt)
#pragma unroll
            for (int e = 0; e < 8; ++e) acc[mt][nt][e] = 0.0f;

    const int n0 = split * 2048 + wave * 256;
    const half_t* aB0 = kexp + (size_t)(head * 32 + l15) * NTOK + n0 + hi * 8;
    const half_t* bB0 = v16  + (size_t)(head * 32 + l15) * NTOK + n0 + hi * 8;

    for (int kk = 0; kk < 256; kk += 32) {
        v16h af[2], bf[2];
#pragma unroll
        for (int mt = 0; mt < 2; ++mt) {   // A = kexp rows (d), K split 8/8
            const half_t* ap = aB0 + (size_t)(mt * 16) * NTOK + kk;
            v8h lo = *(const v8h*)ap;
            v8h hh = *(const v8h*)(ap + 16);
#pragma unroll
            for (int i = 0; i < 8; ++i) { af[mt][i] = lo[i]; af[mt][i + 8] = hh[i]; }
        }
#pragma unroll
        for (int nt = 0; nt < 2; ++nt) {   // B = v rows (e), K split 8/8
            const half_t* bp = bB0 + (size_t)(nt * 16) * NTOK + kk;
            v8h lo = *(const v8h*)bp;
            v8h hh = *(const v8h*)(bp + 16);
#pragma unroll
            for (int i = 0; i < 8; ++i) { bf[nt][i] = lo[i]; bf[nt][i + 8] = hh[i]; }
        }
#pragma unroll
        for (int mt = 0; mt < 2; ++mt)
#pragma unroll
            for (int nt = 0; nt < 2; ++nt)
                acc[mt][nt] = __builtin_amdgcn_wmma_f32_16x16x32_f16(
                    false, af[mt], false, bf[nt], (short)0, acc[mt][nt], false, false);
    }

    float* slot = ctxp + (size_t)(head * 64 + split * 8 + wave) * 1024;
#pragma unroll
    for (int mt = 0; mt < 2; ++mt)
#pragma unroll
        for (int nt = 0; nt < 2; ++nt)
#pragma unroll
            for (int r = 0; r < 8; ++r) {
                int d = mt * 16 + r + (hi ? 8 : 0);
                int e = nt * 16 + l15;
                slot[d * 32 + e] = acc[mt][nt][r];
            }
}

// ---------------- fold: G[o,c] = sum_e Wout[o,h*32+e] * ctx_h[d][e] ----------
__global__ __launch_bounds__(256) void fold_kernel(
    const float* __restrict__ ctxp, const float* __restrict__ rowsum,
    const float* __restrict__ wout, half_t* __restrict__ G16)
{
    __shared__ float ctx[4096];           // [h][d][e]
    const int tid = threadIdx.x;
#pragma unroll
    for (int u = 0; u < 16; ++u) {
        int idx = tid + u * 256;          // 0..4095
        int h = idx >> 10, de = idx & 1023, d = de >> 5;
        float s = 0.0f;
        for (int p = 0; p < 64; ++p) s += ctxp[(size_t)(h * 64 + p) * 1024 + de];
        ctx[idx] = s / rowsum[h * 32 + d];
    }
    __syncthreads();
#pragma unroll
    for (int u = 0; u < 64; ++u) {
        int idx = tid + u * 256;          // 0..16383
        int o = idx >> 7, c = idx & 127;
        int h = c >> 5, d = c & 31;
        float s = 0.0f;
        for (int e = 0; e < 32; ++e)
            s += wout[o * 128 + h * 32 + e] * ctx[h * 1024 + d * 32 + e];
        G16[idx] = (half_t)s;
    }
}

// ---------------------------------------------------------------------------
extern "C" void kernel_launch(void* const* d_in, const int* in_sizes, int n_in,
                              void* d_out, int out_size, void* d_ws, size_t ws_size,
                              hipStream_t stream) {
    const float* x     = (const float*)d_in[0];   // [16,128,128,128]
    const float* w_qkv = (const float*)d_in[1];   // [384,128]
    const float* w_out = (const float*)d_in[2];   // [128,128]
    const float* b_out = (const float*)d_in[3];   // [128]
    float* out = (float*)d_out;                   // [16,128,128,128]

    // workspace layout (all offsets 256B aligned), total ~33.1 MiB
    char* w = (char*)d_ws;
    size_t off = 0;
    half_t* w16    = (half_t*)(w + off); off += (size_t)QK_M * CH * 2;        // 98304
    float*  qkv    = (float*) (w + off); off += (size_t)QK_M * NTOK * 4;      // 25.2 MB
    half_t* kexp   = (half_t*)(w + off); off += (size_t)CH * NTOK * 2;        // 4.2 MB
    half_t* v16    = (half_t*)(w + off); off += (size_t)CH * NTOK * 2;        // 4.2 MB
    float*  rowsum = (float*) (w + off); off += 512;
    float*  ctxp   = (float*) (w + off); off += (size_t)NHEAD * 64 * 1024 * 4;// 1 MB
    half_t* G16    = (half_t*)(w + off); off += (size_t)CH * CH * 2;

    // once: w_qkv -> f16
    convert_f32_to_f16_kernel<<<(QK_M * CH + 255) / 256, 256, 0, stream>>>(
        w_qkv, w16, QK_M * CH);

    for (int b = 0; b < NBATCH; ++b) {
        const float* xb = x + (size_t)b * CH * NTOK;
        float* outb = out + (size_t)b * CH * NTOK;

        // 1) qkv GEMM: [384x128] @ [128x16384]
        gemm_wmma_kernel<<<dim3(NTOK / 128, QK_M / 64), 256, 0, stream>>>(
            w16, xb, qkv, v16, nullptr, QK_M, NTOK, 0, SCALE_Q);

        // 2) softmax stats over k rows (qkv rows 128..255)
        softmax_rows_kernel<<<CH, 256, 0, stream>>>(
            qkv + (size_t)CH * NTOK, kexp, rowsum);

        // 3) context partials
        context_kernel<<<NHEAD * 8, 256, 0, stream>>>(kexp, v16, ctxp);

        // 4) fold into G (includes softmax normalization and W_out)
        fold_kernel<<<1, 256, 0, stream>>>(ctxp, rowsum, w_out, G16);

        // 5) out GEMM: [128x128] @ q[128x16384] + bias
        gemm_wmma_kernel<<<dim3(NTOK / 128, CH / 64), 256, 0, stream>>>(
            G16, qkv /* q rows 0..127 */, outb, nullptr, b_out, CH, NTOK, 1, 0.0f);
    }
    (void)in_sizes; (void)n_in; (void)out_size; (void)ws_size;
}
